// HRoPEEncoderBlock_10153302687755
// MI455X (gfx1250) — matrix operand order repaired and checked by reading, not compiled, not run on hardware
//
#include <hip/hip_runtime.h>
#include <hip/hip_bf16.h>
#include <math.h>

// ---------------- types ----------------
typedef __bf16 bf16_t;
typedef __attribute__((ext_vector_type(16))) __bf16 v16bf;
typedef __attribute__((ext_vector_type(8)))  __bf16 v8bf;
typedef __attribute__((ext_vector_type(8)))  float  v8f;

#define B_   2
#define S_   2048
#define D_   1024
#define H_   16
#define HD_  64
#define HID_ 4096
#define M_   (B_ * S_)            // 4096 rows

__device__ __forceinline__ v16bf make_frag(v8bf a, v8bf b) {
    return __builtin_shufflevector(a, b, 0,1,2,3,4,5,6,7,8,9,10,11,12,13,14,15);
}
__device__ __forceinline__ v8f wmma_bf16(v16bf a, v16bf b, v8f c) {
    return __builtin_amdgcn_wmma_f32_16x16x32_bf16(false, a, false, b, (short)0, c, false, false);
}

// ---------------- weight convert + transpose: W[K][N] f32 -> Wt[N][K] bf16 ----
__global__ __launch_bounds__(256)
void wt_transpose_kernel(const float* __restrict__ W, bf16_t* __restrict__ Wt,
                         int K, int N) {
    __shared__ float tile[32][33];
    const int k0 = blockIdx.y * 32;
    const int n0 = blockIdx.x * 32;
    const int tx = threadIdx.x & 31;
    const int ty = threadIdx.x >> 5;          // 0..7
#pragma unroll
    for (int j = 0; j < 32; j += 8)
        tile[ty + j][tx] = W[(size_t)(k0 + ty + j) * N + n0 + tx];
    __syncthreads();
#pragma unroll
    for (int j = 0; j < 32; j += 8)
        Wt[(size_t)(n0 + ty + j) * K + k0 + tx] = (bf16_t)tile[tx][ty + j];
}

// ---------------- RMSNorm row kernel: f32 (rowsxD) -> bf16 ----------------
__global__ __launch_bounds__(256)
void rmsnorm_kernel(const float* __restrict__ x, const float* __restrict__ w,
                    bf16_t* __restrict__ out) {
    const int row  = blockIdx.x;
    const int tid  = threadIdx.x;
    const int lane = tid & 31;
    const int wv   = tid >> 5;
    const float* xr = x + (size_t)row * D_;
    float ss = 0.f;
#pragma unroll
    for (int i = tid; i < D_; i += 256) { float t = xr[i]; ss += t * t; }
#pragma unroll
    for (int off = 16; off >= 1; off >>= 1) ss += __shfl_xor(ss, off, 32);
    __shared__ float red[8];
    if (lane == 0) red[wv] = ss;
    __syncthreads();
    float tot = 0.f;
#pragma unroll
    for (int i = 0; i < 8; ++i) tot += red[i];
    const float scale = rsqrtf(tot * (1.0f / (float)D_) + 1e-6f);
#pragma unroll
    for (int i = tid; i < D_; i += 256)
        out[(size_t)row * D_ + i] = (bf16_t)(xr[i] * scale * w[i]);
}

// ---------------- per-head RMSNorm + RoPE : f32 -> bf16 ----------------
// 128 threads = 4 waves; each wave handles one (b,s,h) slice;
// lane t handles dims t and t+32 (the rope rotation pair).
__global__ __launch_bounds__(128)
void qk_rope_kernel(const float* __restrict__ qk, const float* __restrict__ hw,
                    const int* __restrict__ positions, bf16_t* __restrict__ out) {
    const int idx = blockIdx.x * 4 + (threadIdx.x >> 5);  // ((b*S + s)*H + h)
    const int h  = idx & (H_ - 1);
    const int bs = idx >> 4;                  // b*S + s
    const int t  = threadIdx.x & 31;          // 0..31
    const float* vr = qk + (size_t)bs * D_ + h * HD_;
    float x1 = vr[t], x2 = vr[t + 32];
    float ss = x1 * x1 + x2 * x2;
#pragma unroll
    for (int off = 16; off >= 1; off >>= 1) ss += __shfl_xor(ss, off, 32);
    const float scale = rsqrtf(ss * (1.0f / (float)HD_) + 1e-6f);
    x1 *= scale * hw[t];
    x2 *= scale * hw[t + 32];
    const float pos = (float)positions[bs];
    // inv_freq = 10000^(-t/32) = exp(-t * ln(10000)/32)
    const float fr = pos * __expf(-(float)t * (9.210340371976184f / 32.0f));
    const float c = __cosf(fr), sn = __sinf(fr);
    bf16_t* o = out + (size_t)bs * D_ + h * HD_;
    o[t]      = (bf16_t)(x1 * c - x2 * sn);
    o[t + 32] = (bf16_t)(x1 * sn + x2 * c);
}

// ---------------- WMMA GEMM: out(MxN) = A(MxK,bf16) x Wt(NxK,bf16)^T ------
// block: 256 thr = 8 waves in 2(M)x4(N); block tile 64x128; wave tile 32x32
// MODE 0: outF = acc            MODE 1: outF = resid + acc
// MODE 2: outB = gelu(acc)      MODE 3: outB = (bf16)acc
// Double-buffered LDS A staging: one barrier per K-step, next tile's
// global_load/ds_store issued before the current tile's WMMAs.
template <int MODE>
__global__ __launch_bounds__(256)
void gemm_wmma_kernel(const bf16_t* __restrict__ A, const bf16_t* __restrict__ Wt,
                      const float* __restrict__ resid, float* __restrict__ outF,
                      bf16_t* __restrict__ outB, int M, int N, int K) {
    __shared__ bf16_t As[2][64][32];          // 2 x 4 KB
    const int tid   = threadIdx.x;
    const int lane  = tid & 31;
    const int wave  = tid >> 5;
    const int col16 = lane & 15;
    const int hb    = lane >> 4;
    const int wm    = wave >> 2;              // 0..1
    const int wn    = wave & 3;               // 0..3
    const int m0    = blockIdx.y * 64;
    const int n0    = blockIdx.x * 128;

    v8f acc[2][2];
#pragma unroll
    for (int i = 0; i < 2; ++i)
#pragma unroll
        for (int j = 0; j < 2; ++j)
            acc[i][j] = (v8f){0.f,0.f,0.f,0.f,0.f,0.f,0.f,0.f};

    const int lr = tid >> 2;                  // 0..63 : A-stage row
    const int lc = (tid & 3) << 3;            // 0,8,16,24
    const bf16_t* aRow = A + (size_t)(m0 + lr) * K + lc;

    // preload k0 = 0 into buffer 0
    *(v8bf*)&As[0][lr][lc] = *(const v8bf*)aRow;
    __syncthreads();

    for (int k0 = 0; k0 < K; k0 += 32) {
        const int buf = (k0 >> 5) & 1;
        // stage next tile into the other buffer while we compute this one
        if (k0 + 32 < K)
            *(v8bf*)&As[buf ^ 1][lr][lc] = *(const v8bf*)(aRow + k0 + 32);
        if (k0 + 64 < K)
            __builtin_prefetch(aRow + k0 + 64, 0, 1);

        v16bf afrag[2];
#pragma unroll
        for (int mt = 0; mt < 2; ++mt) {
            const int row = wm * 32 + mt * 16 + col16;
            v8bf a0 = *(const v8bf*)&As[buf][row][hb * 8];
            v8bf a1 = *(const v8bf*)&As[buf][row][16 + hb * 8];
            afrag[mt] = make_frag(a0, a1);
        }
        v16bf bfrag[2];
#pragma unroll
        for (int nt = 0; nt < 2; ++nt) {
            const int col = n0 + wn * 32 + nt * 16 + col16;
            const bf16_t* p = Wt + (size_t)col * K + k0 + hb * 8;
            bfrag[nt] = make_frag(*(const v8bf*)p, *(const v8bf*)(p + 16));
        }
#pragma unroll
        for (int mt = 0; mt < 2; ++mt)
#pragma unroll
            for (int nt = 0; nt < 2; ++nt)
                acc[mt][nt] = wmma_bf16(afrag[mt], bfrag[nt], acc[mt][nt]);
        __syncthreads();
    }

    // straight-line epilogue (MODE is compile-time)
#pragma unroll
    for (int mt = 0; mt < 2; ++mt)
#pragma unroll
        for (int nt = 0; nt < 2; ++nt) {
            const int rbase = m0 + wm * 32 + mt * 16 + 8 * hb;
            const int c     = n0 + wn * 32 + nt * 16 + col16;
#pragma unroll
            for (int i = 0; i < 8; ++i) {
                const size_t o = (size_t)(rbase + i) * N + c;
                const float v = acc[mt][nt][i];
                if (MODE == 0) {
                    outF[o] = v;
                } else if (MODE == 1) {
                    outF[o] = resid[o] + v;
                } else if (MODE == 2) {   // exact GELU -> bf16
                    outB[o] = (bf16_t)(0.5f * v * (1.0f + erff(v * 0.70710678118654752f)));
                } else {                  // plain bf16
                    outB[o] = (bf16_t)v;
                }
            }
        }
}

// ---------------- flash-style ReLU^2 causal attention ----------------
// grid (B*H, S/64); block 128 thr = 4 waves; each wave owns 16 query rows.
__global__ __launch_bounds__(128)
void attn_kernel(const bf16_t* __restrict__ q, const bf16_t* __restrict__ k,
                 const bf16_t* __restrict__ v, bf16_t* __restrict__ out) {
    __shared__ bf16_t Kb[32][64];             // key block  (row-major, d contig)
    __shared__ bf16_t VT[64][32];             // V^T        (d-major, key contig)
    __shared__ bf16_t Wb[64][32];             // relu^2 scores, bf16
    const int bh   = blockIdx.x;
    const int b    = bh >> 4;
    const int h    = bh & (H_ - 1);
    const int qb0  = blockIdx.y * 64;
    const int tid  = threadIdx.x;
    const int lane = tid & 31;
    const int wave = tid >> 5;                // 0..3
    const int col16 = lane & 15;
    const int hb    = lane >> 4;

    // Q A-fragments straight from global (row = query, d contiguous)
    const int qrowA = qb0 + wave * 16 + col16;
    const bf16_t* qp = q + ((size_t)(b * S_ + qrowA)) * D_ + h * HD_;
    v16bf qfrag[2];
#pragma unroll
    for (int ks = 0; ks < 2; ++ks) {
        v8bf a0 = *(const v8bf*)(qp + ks * 32 + hb * 8);
        v8bf a1 = *(const v8bf*)(qp + ks * 32 + 16 + hb * 8);
        qfrag[ks] = make_frag(a0, a1);
    }

    v8f acc[4];
#pragma unroll
    for (int i = 0; i < 4; ++i) acc[i] = (v8f){0.f,0.f,0.f,0.f,0.f,0.f,0.f,0.f};
    float dsum[8] = {0.f,0.f,0.f,0.f,0.f,0.f,0.f,0.f};

    const int nkb = (qb0 + 64) >> 5;          // causal: keys <= qb0+63
    const int lr  = tid >> 2;                 // 0..31 K-staging row
    const int lc  = (tid & 3) << 4;           // 0,16,32,48
    const int kp  = tid & 15;                 // V-staging: keys 2kp, 2kp+1
    const int dc  = (tid >> 4) << 3;          // V-staging: d chunk 0..56 step 8

    for (int kb = 0; kb < nkb; ++kb) {
        const int kpos0 = kb << 5;
        // stage K block (row-major)
        {
            const bf16_t* src = k + ((size_t)(b * S_ + kpos0 + lr)) * D_ + h * HD_ + lc;
            *(v8bf*)&Kb[lr][lc]     = *(const v8bf*)src;
            *(v8bf*)&Kb[lr][lc + 8] = *(const v8bf*)(src + 8);
        }
        // stage V transposed: pack key-pairs into dword DS stores
        {
            const bf16_t* sv = v + ((size_t)(b * S_ + kpos0 + 2 * kp)) * D_ + h * HD_ + dc;
            v8bf r0 = *(const v8bf*)sv;
            v8bf r1 = *(const v8bf*)(sv + D_);
#pragma unroll
            for (int j = 0; j < 8; ++j) {
                union { bf16_t e[2]; unsigned u; } pk;
                pk.e[0] = r0[j];
                pk.e[1] = r1[j];
                *(unsigned*)&VT[dc + j][2 * kp] = pk.u;
            }
        }
        __syncthreads();

        // scores (16 queries x 32 keys), K-dim = 64
#pragma unroll
        for (int nt = 0; nt < 2; ++nt) {
            v8f c8 = (v8f){0.f,0.f,0.f,0.f,0.f,0.f,0.f,0.f};
#pragma unroll
            for (int ks = 0; ks < 2; ++ks) {
                const bf16_t* p = &Kb[nt * 16 + col16][ks * 32 + hb * 8];
                v16bf bfrag = make_frag(*(const v8bf*)p, *(const v8bf*)(p + 16));
                c8 = wmma_bf16(qfrag[ks], bfrag, c8);
            }
            const int keycol = kpos0 + nt * 16 + col16;
#pragma unroll
            for (int i = 0; i < 8; ++i) {
                const int qrow = qb0 + wave * 16 + i + 8 * hb;
                float s = c8[i] * 0.125f;                 // 1/sqrt(64)
                s = (keycol <= qrow && s > 0.f) ? s * s : 0.f;   // relu^2 + causal
                dsum[i] += s;
                Wb[wave * 16 + i + 8 * hb][nt * 16 + col16] = (bf16_t)s;
            }
        }

        // P (16x32) x V (32x64) accumulate
        {
            const bf16_t* wp = &Wb[wave * 16 + col16][0];
            v16bf afrag = make_frag(*(const v8bf*)(wp + hb * 8),
                                    *(const v8bf*)(wp + 16 + hb * 8));
#pragma unroll
            for (int nt2 = 0; nt2 < 4; ++nt2) {
                const bf16_t* p = &VT[nt2 * 16 + col16][hb * 8];
                v16bf bfrag = make_frag(*(const v8bf*)p, *(const v8bf*)(p + 16));
                acc[nt2] = wmma_bf16(afrag, bfrag, acc[nt2]);
            }
        }
        __syncthreads();
    }

    // reduce denominator across the 16 lanes that share each row
#pragma unroll
    for (int i = 0; i < 8; ++i) {
        float s = dsum[i];
#pragma unroll
        for (int off = 8; off >= 1; off >>= 1) s += __shfl_xor(s, off, 16);
        dsum[i] = 1.0f / fmaxf(s, 1e-6f);
    }

    // write attn output (b, s, h*64+d) as bf16
#pragma unroll
    for (int nt2 = 0; nt2 < 4; ++nt2)
#pragma unroll
        for (int i = 0; i < 8; ++i) {
            const int qrow = qb0 + wave * 16 + i + 8 * hb;
            const int d    = nt2 * 16 + col16;
            out[((size_t)(b * S_ + qrow)) * D_ + h * HD_ + d] =
                (bf16_t)(acc[nt2][i] * dsum[i]);
        }
}

// ---------------- host launch ----------------
#define MiB (1ull << 20)

extern "C" void kernel_launch(void* const* d_in, const int* in_sizes, int n_in,
                              void* d_out, int out_size, void* d_ws, size_t ws_size,
                              hipStream_t stream) {
    const float* x      = (const float*)d_in[0];
    const int*   pos    = (const int*)  d_in[1];
    // d_in[2] attn_mask: known causal tril, unused
    const float* n1w    = (const float*)d_in[3];
    const float* n2w    = (const float*)d_in[4];
    const float* qnw    = (const float*)d_in[5];
    const float* knw    = (const float*)d_in[6];
    const float* Wq     = (const float*)d_in[7];
    const float* Wk     = (const float*)d_in[8];
    const float* Wv     = (const float*)d_in[9];
    const float* Wo     = (const float*)d_in[10];
    const float* W1     = (const float*)d_in[11];
    const float* W2     = (const float*)d_in[12];
    float* out          = (float*)d_out;

    char* ws = (char*)d_ws;
    bf16_t* hB    = (bf16_t*)(ws + 0 * MiB);    // 8 MiB  rmsnorm1(x)
    bf16_t* WtQ   = (bf16_t*)(ws + 8 * MiB);    // 2
    bf16_t* WtK   = (bf16_t*)(ws + 10 * MiB);   // 2
    bf16_t* WtV   = (bf16_t*)(ws + 12 * MiB);   // 2
    bf16_t* WtO   = (bf16_t*)(ws + 14 * MiB);   // 2
    bf16_t* Wt1   = (bf16_t*)(ws + 16 * MiB);   // 8
    bf16_t* Wt2   = (bf16_t*)(ws + 24 * MiB);   // 8
    float*  qF    = (float*) (ws + 32 * MiB);   // 16 (later aliased as x1)
    float*  kF    = (float*) (ws + 48 * MiB);   // 16 (later aliased as gelu, 32 MiB)
    bf16_t* qB    = (bf16_t*)(ws + 80 * MiB);   // 8
    bf16_t* kB    = (bf16_t*)(ws + 88 * MiB);   // 8
    bf16_t* vB    = (bf16_t*)(ws + 96 * MiB);   // 8
    bf16_t* attnB = (bf16_t*)(ws + 104 * MiB);  // 8 (later aliased as h2)
    float*  x1F   = qF;                         // alias: q_f32 dead after rope
    bf16_t* gB    = (bf16_t*)kF;                // alias: 32 MiB gelu output
    bf16_t* h2B   = attnB;                      // alias: attn dead after Wo gemm

    // 1) weights -> bf16, transposed [N][K]
    wt_transpose_kernel<<<dim3(D_/32,   D_/32), 256, 0, stream>>>(Wq, WtQ, D_,   D_);
    wt_transpose_kernel<<<dim3(D_/32,   D_/32), 256, 0, stream>>>(Wk, WtK, D_,   D_);
    wt_transpose_kernel<<<dim3(D_/32,   D_/32), 256, 0, stream>>>(Wv, WtV, D_,   D_);
    wt_transpose_kernel<<<dim3(D_/32,   D_/32), 256, 0, stream>>>(Wo, WtO, D_,   D_);
    wt_transpose_kernel<<<dim3(HID_/32, D_/32), 256, 0, stream>>>(W1, Wt1, D_,   HID_);
    wt_transpose_kernel<<<dim3(D_/32, HID_/32), 256, 0, stream>>>(W2, Wt2, HID_, D_);

    // 2) h = rmsnorm(x, norm1_w) -> bf16
    rmsnorm_kernel<<<M_, 256, 0, stream>>>(x, n1w, hB);

    // 3) q/k projections (f32 out, rope follows); v projection straight to bf16
    const dim3 gD(D_ / 128, M_ / 64);
    gemm_wmma_kernel<0><<<gD, 256, 0, stream>>>(hB, WtQ, nullptr, qF, nullptr, M_, D_, D_);
    gemm_wmma_kernel<0><<<gD, 256, 0, stream>>>(hB, WtK, nullptr, kF, nullptr, M_, D_, D_);
    gemm_wmma_kernel<3><<<gD, 256, 0, stream>>>(hB, WtV, nullptr, nullptr, vB, M_, D_, D_);

    // 4) per-head norm + rope (q,k) -> bf16
    qk_rope_kernel<<<(M_ * H_) / 4, 128, 0, stream>>>(qF, qnw, pos, qB);
    qk_rope_kernel<<<(M_ * H_) / 4, 128, 0, stream>>>(kF, knw, pos, kB);

    // 5) attention
    attn_kernel<<<dim3(B_ * H_, S_ / 64), 128, 0, stream>>>(qB, kB, vB, attnB);

    // 6) x1 = x + attn @ Wo
    gemm_wmma_kernel<1><<<gD, 256, 0, stream>>>(attnB, WtO, x, x1F, nullptr, M_, D_, D_);

    // 7) h2 = rmsnorm(x1, norm2_w)
    rmsnorm_kernel<<<M_, 256, 0, stream>>>(x1F, n2w, h2B);

    // 8) g = gelu(h2 @ W1) -> bf16
    gemm_wmma_kernel<2><<<dim3(HID_ / 128, M_ / 64), 256, 0, stream>>>(
        h2B, Wt1, nullptr, nullptr, gB, M_, HID_, D_);

    // 9) out = x1 + g @ W2
    gemm_wmma_kernel<1><<<gD, 256, 0, stream>>>(gB, Wt2, x1F, out, nullptr, M_, D_, HID_);
}
